// HungarianMatcher_89197880803896
// MI455X (gfx1250) — matrix operand order repaired and checked by reading, not compiled, not run on hardware
//
#include <hip/hip_runtime.h>
#include <hip/hip_bf16.h>

typedef _Float16 h8   __attribute__((ext_vector_type(8)));
typedef _Float16 v16h __attribute__((ext_vector_type(16)));
typedef float    v8f  __attribute__((ext_vector_type(8)));

#define B_   4
#define Q_   300
#define NC   81
#define HO   160
#define WO   160
#define PP   (HO * WO)          // 25600 pixels per mask
#define MT   400                // targets
#define NT   (B_ * Q_)          // 1200 predictions
#define HM   256
#define WM   256

static __device__ __forceinline__ int clampi(int v, int lo, int hi) {
    return v < lo ? lo : (v > hi ? hi : v);
}

// ---------------- Stage 0: bilinear resize tgt masks 256->160 (fp16) + row sums ----
__global__ __launch_bounds__(256) void resize_tgt_kernel(
    const float* __restrict__ tgt_masks, _Float16* __restrict__ t_h,
    float* __restrict__ tsum) {
    __shared__ float red[256];
    const int m = blockIdx.x;
    const float* src = tgt_masks + (size_t)m * (HM * WM);
    float acc = 0.f;
    for (int p = threadIdx.x; p < PP; p += 256) {
        const int y = p / WO, x = p - y * WO;
        const float sy = (y + 0.5f) * ((float)HM / (float)HO) - 0.5f;
        const float sx = (x + 0.5f) * ((float)WM / (float)WO) - 0.5f;
        const float fy0 = floorf(sy), fx0 = floorf(sx);
        const float wy = sy - fy0, wx = sx - fx0;
        const int y0 = clampi((int)fy0, 0, HM - 1);
        const int y1 = clampi((int)fy0 + 1, 0, HM - 1);
        const int x0 = clampi((int)fx0, 0, WM - 1);
        const int x1 = clampi((int)fx0 + 1, 0, WM - 1);
        const float v00 = src[y0 * WM + x0], v01 = src[y0 * WM + x1];
        const float v10 = src[y1 * WM + x0], v11 = src[y1 * WM + x1];
        const float v = (1.f - wy) * ((1.f - wx) * v00 + wx * v01)
                      +        wy  * ((1.f - wx) * v10 + wx * v11);
        t_h[(size_t)m * PP + p] = (_Float16)v;
        acc += v;
    }
    red[threadIdx.x] = acc;
    __syncthreads();
    for (int s = 128; s > 0; s >>= 1) {
        if ((int)threadIdx.x < s) red[threadIdx.x] += red[threadIdx.x + s];
        __syncthreads();
    }
    if (threadIdx.x == 0) tsum[m] = red[0];
}

// ---------------- Stage 1: stage x and sigmoid(x) as fp16, per-row sums ------------
__global__ __launch_bounds__(256) void prep_pred_kernel(
    const float* __restrict__ pred_masks,
    _Float16* __restrict__ x_h, _Float16* __restrict__ s_h,
    float* __restrict__ slog1mp, float* __restrict__ psum) {
    __shared__ float r0[256];
    __shared__ float r1[256];
    const int n = blockIdx.x;
    const float* src = pred_masks + (size_t)n * PP;
    float a0 = 0.f, a1 = 0.f;
    for (int p = threadIdx.x; p < PP; p += 256) {
        const float x  = src[p];
        const float sp = fmaxf(x, 0.f) + log1pf(__expf(-fabsf(x))); // softplus(x)
        const float sg = 1.f / (1.f + __expf(-x));
        x_h[(size_t)n * PP + p] = (_Float16)x;
        s_h[(size_t)n * PP + p] = (_Float16)sg;
        a0 -= sp;   // sum over p of log sigmoid(-x)
        a1 += sg;   // sum over p of sigmoid(x)
    }
    r0[threadIdx.x] = a0;
    r1[threadIdx.x] = a1;
    __syncthreads();
    for (int s = 128; s > 0; s >>= 1) {
        if ((int)threadIdx.x < s) {
            r0[threadIdx.x] += r0[threadIdx.x + s];
            r1[threadIdx.x] += r1[threadIdx.x + s];
        }
        __syncthreads();
    }
    if (threadIdx.x == 0) { slog1mp[n] = r0[0]; psum[n] = r1[0]; }
}

// ---------------- Stage 2: softmax class probabilities -----------------------------
__global__ __launch_bounds__(256) void class_prob_kernel(
    const float* __restrict__ logits, float* __restrict__ prob) {
    const int n = blockIdx.x * blockDim.x + threadIdx.x;
    if (n >= NT) return;
    const float* l = logits + (size_t)n * NC;
    float mx = -3.4e38f;
    for (int c = 0; c < NC; ++c) mx = fmaxf(mx, l[c]);
    float s = 0.f;
    for (int c = 0; c < NC; ++c) s += __expf(l[c] - mx);
    const float inv = 1.f / s;
    for (int c = 0; c < NC; ++c) prob[(size_t)n * NC + c] = __expf(l[c] - mx) * inv;
}

// ---------------- Stage 3: dual-GEMM via v_wmma_f32_16x16x32_f16 + fused epilogue --
// Each wave computes one 16(N) x 16(M) output tile; two accumulators:
//   S1 = x @ t^T  (mask BCE via log-sigmoid identity),  S2 = sigmoid(x) @ t^T (dice)
__global__ __launch_bounds__(160) void wmma_cost_kernel(
    const _Float16* __restrict__ x_h, const _Float16* __restrict__ s_h,
    const _Float16* __restrict__ t_h,
    const float* __restrict__ slog1mp, const float* __restrict__ psum,
    const float* __restrict__ tsum, const float* __restrict__ prob,
    const float* __restrict__ pred_boxes, const float* __restrict__ tgt_boxes,
    const int* __restrict__ tgt_ids, float* __restrict__ out) {
    const int lane  = threadIdx.x & 31;
    const int wave  = threadIdx.x >> 5;
    const int n0    = (blockIdx.y * 5 + wave) * 16;   // 75 N-tiles = 15 blocks * 5 waves
    const int m0    = blockIdx.x * 16;                // 25 M-tiles
    const int lmod  = lane & 15;
    const int lhalf = lane >> 4;

    // CDNA5 16-bit A layout: lane = row (M), K = lhalf*8 + {0..7} and 16+lhalf*8+{0..7}
    // CDNA5 16-bit B layout: lane = col (N), K = lhalf*8 + {0..7} and 16+lhalf*8+{0..7}
    const _Float16* ax = x_h + (size_t)(n0 + lmod) * PP + lhalf * 8;
    const _Float16* as = s_h + (size_t)(n0 + lmod) * PP + lhalf * 8;
    const _Float16* bt = t_h + (size_t)(m0 + lmod) * PP + lhalf * 8;

    v8f acc1 = {0.f, 0.f, 0.f, 0.f, 0.f, 0.f, 0.f, 0.f};
    v8f acc2 = {0.f, 0.f, 0.f, 0.f, 0.f, 0.f, 0.f, 0.f};

    for (int k0 = 0; k0 < PP; k0 += 32) {
        __builtin_prefetch(ax + k0 + 256, 0, 1);   // global_prefetch_b8
        __builtin_prefetch(as + k0 + 256, 0, 1);
        __builtin_prefetch(bt + k0 + 256, 0, 1);

        const h8 aL0 = *(const h8*)(ax + k0);
        const h8 aL1 = *(const h8*)(ax + k0 + 16);
        const h8 sL0 = *(const h8*)(as + k0);
        const h8 sL1 = *(const h8*)(as + k0 + 16);
        const h8 bL0 = *(const h8*)(bt + k0);
        const h8 bL1 = *(const h8*)(bt + k0 + 16);

        const v16h A  = __builtin_shufflevector(aL0, aL1, 0,1,2,3,4,5,6,7,8,9,10,11,12,13,14,15);
        const v16h S  = __builtin_shufflevector(sL0, sL1, 0,1,2,3,4,5,6,7,8,9,10,11,12,13,14,15);
        const v16h Bm = __builtin_shufflevector(bL0, bL1, 0,1,2,3,4,5,6,7,8,9,10,11,12,13,14,15);

        acc1 = __builtin_amdgcn_wmma_f32_16x16x32_f16(false, A, false, Bm, (short)0, acc1, false, false);
        acc2 = __builtin_amdgcn_wmma_f32_16x16x32_f16(false, S, false, Bm, (short)0, acc2, false, false);
    }

    // Epilogue: C/D layout -> lane holds column m = m0 + lane%16; VGPR v holds row v + 8*lhalf.
    const int   m   = m0 + lmod;
    const float4 tb = *(const float4*)(tgt_boxes + m * 4);
    const int   cid = tgt_ids[m];
    const float tsm = tsum[m];
    const float tx0 = tb.x - 0.5f * tb.z, ty0 = tb.y - 0.5f * tb.w;
    const float tx1 = tb.x + 0.5f * tb.z, ty1 = tb.y + 0.5f * tb.w;
    const float areat = tb.z * tb.w;

#pragma unroll
    for (int v = 0; v < 8; ++v) {
        const int n = n0 + v + 8 * lhalf;
        const float4 pb = *(const float4*)(pred_boxes + n * 4);
        // L1 box cost (cxcywh space)
        const float cb = fabsf(pb.x - tb.x) + fabsf(pb.y - tb.y)
                       + fabsf(pb.z - tb.z) + fabsf(pb.w - tb.w);
        // GIoU in xyxy space
        const float px0 = pb.x - 0.5f * pb.z, py0 = pb.y - 0.5f * pb.w;
        const float px1 = pb.x + 0.5f * pb.z, py1 = pb.y + 0.5f * pb.w;
        const float areap = pb.z * pb.w;
        const float ix0 = fmaxf(px0, tx0), iy0 = fmaxf(py0, ty0);
        const float ix1 = fminf(px1, tx1), iy1 = fminf(py1, ty1);
        const float inter = fmaxf(ix1 - ix0, 0.f) * fmaxf(iy1 - iy0, 0.f);
        const float uni   = areap + areat - inter;
        const float iou   = inter / uni;
        const float ex0 = fminf(px0, tx0), ey0 = fminf(py0, ty0);
        const float ex1 = fmaxf(px1, tx1), ey1 = fmaxf(py1, ty1);
        const float areae = fmaxf(ex1 - ex0, 0.f) * fmaxf(ey1 - ey0, 0.f);
        const float giou  = iou - (areae - uni) / areae;

        const float cclass = -prob[(size_t)n * NC + cid];
        const float cmask  = -(acc1[v] + slog1mp[n]) * (1.f / (float)PP);
        const float cdice  = 1.f - (2.f * acc2[v] + 1e-5f) / (psum[n] + tsm + 1e-5f);

        out[(size_t)n * MT + m] =
            5.f * cb + 2.f * (-giou) + 2.f * cclass + 5.f * cmask + 5.f * cdice;
    }
}

extern "C" void kernel_launch(void* const* d_in, const int* in_sizes, int n_in,
                              void* d_out, int out_size, void* d_ws, size_t ws_size,
                              hipStream_t stream) {
    const float* pred_logits = (const float*)d_in[0];   // [4,300,81]
    const float* pred_boxes  = (const float*)d_in[1];   // [4,300,4]
    const float* pred_masks  = (const float*)d_in[2];   // [4,300,160,160]
    const float* tgt_boxes   = (const float*)d_in[3];   // [400,4]
    const float* tgt_masks   = (const float*)d_in[4];   // [400,256,256]
    const int*   tgt_ids     = (const int*)d_in[5];     // [400]
    float*       out         = (float*)d_out;           // [4,300,400]

    // Workspace carve-up (256B aligned); total ~143.4 MB, fits in MI455X's 192MB L2.
    char* ws = (char*)d_ws;
    size_t off = 0;
    auto carve = [&](size_t bytes) -> char* {
        char* p = ws + off;
        off = (off + bytes + 255) & ~(size_t)255;
        return p;
    };
    _Float16* t_h     = (_Float16*)carve((size_t)MT * PP * 2);   // resized tgt, fp16
    _Float16* x_h     = (_Float16*)carve((size_t)NT * PP * 2);   // mask logits, fp16
    _Float16* s_h     = (_Float16*)carve((size_t)NT * PP * 2);   // sigmoid(logits), fp16
    float*    tsum    = (float*)carve((size_t)MT * 4);
    float*    slog1mp = (float*)carve((size_t)NT * 4);
    float*    psum    = (float*)carve((size_t)NT * 4);
    float*    prob    = (float*)carve((size_t)NT * NC * 4);

    resize_tgt_kernel<<<MT, 256, 0, stream>>>(tgt_masks, t_h, tsum);
    prep_pred_kernel<<<NT, 256, 0, stream>>>(pred_masks, x_h, s_h, slog1mp, psum);
    class_prob_kernel<<<(NT + 255) / 256, 256, 0, stream>>>(pred_logits, prob);

    dim3 grid(MT / 16, (NT / 16) / 5, 1);   // 25 x 15 blocks, 5 waves each
    wmma_cost_kernel<<<grid, 160, 0, stream>>>(
        x_h, s_h, t_h, slog1mp, psum, tsum, prob,
        pred_boxes, tgt_boxes, tgt_ids, out);
}
